// Lstm_72481868087577
// MI455X (gfx1250) — matrix operand order repaired and checked by reading, not compiled, not run on hardware
//
#include <hip/hip_runtime.h>
#include <hip/hip_bf16.h>
#include <cstdint>

// ---------------------------------------------------------------------------
// Problem constants (from reference): T=128, B=256, V=256, E=512, NN=1024
// ---------------------------------------------------------------------------
#define T_ 128
#define B_ 256
#define V_ 256
#define E_ 512
#define NN_ 1024

typedef __attribute__((ext_vector_type(4)))  int    v4i;
typedef __attribute__((ext_vector_type(8)))  float  v8f;
typedef __attribute__((ext_vector_type(16))) __bf16 v16bf;

union bf16x16 { v4i q[2]; v16bf v; };

__device__ inline __bf16 f2bf(float f) {
    // round-to-nearest-even fp32 -> bf16
    unsigned u = __builtin_bit_cast(unsigned, f);
    unsigned r = u + 0x7FFFu + ((u >> 16) & 1u);
    unsigned short h = (unsigned short)(r >> 16);
    return __builtin_bit_cast(__bf16, h);
}

__device__ inline void store_val(float*  p, float v) { *p = v; }
__device__ inline void store_val(__bf16* p, float v) { *p = f2bf(v); }

// ---------------------------------------------------------------------------
// Conversion / init kernels
// ---------------------------------------------------------------------------
__global__ void cvt_f32_to_bf16(const float* __restrict__ in, __bf16* __restrict__ out, int n) {
    int i = blockIdx.x * blockDim.x + threadIdx.x;
    if (i < n) out[i] = f2bf(in[i]);
}

// in: K x N (row-major fp32)  ->  out: N x K (row-major bf16)
__global__ void cvt_transpose_f32_to_bf16(const float* __restrict__ in, __bf16* __restrict__ out,
                                          int K, int N) {
    int i = blockIdx.x * blockDim.x + threadIdx.x;
    if (i >= K * N) return;
    int n = i / K, k = i % K;
    out[i] = f2bf(in[(size_t)k * N + n]);
}

__global__ void zero_bytes_kernel(unsigned char* __restrict__ p, int n) {
    int i = blockIdx.x * blockDim.x + threadIdx.x;
    if (i < n) p[i] = 0;
}

// ---------------------------------------------------------------------------
// bf16 WMMA GEMM:  out[M,N] = concat(A0[:,0:K0], A1[:,0:K1]) @ Wt^T + bias
//   A0/A1 : row-major bf16 (A1 provides columns K0..K0+K1-1)
//   Wt    : N x K row-major bf16 (pre-transposed weights)
//   out   : fp32 or bf16, row-major M x N
// Tile: 64x128 per workgroup, 128 threads = 4 waves in a 2x2 grid,
//       each wave computes a 32x64 sub-tile as 2x4 WMMA accumulators.
// Double-buffered LDS: one barrier per K-step.
// ---------------------------------------------------------------------------
#define TILE_M 64
#define TILE_N 128
#define TILE_K 32

template<typename OutT>
__global__ __launch_bounds__(128)
void gemm_bf16_wmma(const __bf16* __restrict__ A0, int K0,
                    const __bf16* __restrict__ A1, int K1,
                    const __bf16* __restrict__ Wt,
                    const float*  __restrict__ bias,
                    OutT* __restrict__ out,
                    int M, int N)
{
    const int K = K0 + K1;
    __shared__ __bf16 ldsA[2][TILE_M * TILE_K];   // 2 x 4 KB
    __shared__ __bf16 ldsB[2][TILE_N * TILE_K];   // 2 x 8 KB

    const int tid  = threadIdx.x;
    const int lane = tid & 31;
    const int wave = tid >> 5;
    const int wm   = wave >> 1;        // 0..1  (32-row strip)
    const int wn   = wave & 1;         // 0..1  (64-col strip)

    const int mTile = blockIdx.y * TILE_M;
    const int nTile = blockIdx.x * TILE_N;

    v8f acc[2][4] = {};

    // Loader mapping:
    //   A tile 64x32 = 128 chunks of 16 bf16 -> 1 chunk/thread
    //   B tile 128x32 = 256 chunks           -> 2 chunks/thread (one full row)
    const int aRow = tid >> 1;             // 0..63
    const int aCol = (tid & 1) * 16;       // 0 or 16
    const int bRow = tid;                  // 0..127

    const int half = lane >> 4;            // 0 or 1
    const int l15  = lane & 15;

    const int gmA = mTile + aRow;
    const int gnB = nTile + bRow;

    bf16x16 aReg, bReg0, bReg1;

    // ---- prologue: load K-tile 0 into registers ----
    {
        const int gk = aCol;
        const __bf16* src = (gk < K0) ? (A0 + (size_t)gmA * K0 + gk)
                                      : (A1 + (size_t)gmA * K1 + (gk - K0));
        aReg  = *(const bf16x16*)src;
        bReg0 = *(const bf16x16*)(Wt + (size_t)gnB * K + 0);
        bReg1 = *(const bf16x16*)(Wt + (size_t)gnB * K + 16);
    }
    *(bf16x16*)&ldsA[0][aRow * TILE_K + aCol] = aReg;
    *(bf16x16*)&ldsB[0][bRow * TILE_K +    0] = bReg0;
    *(bf16x16*)&ldsB[0][bRow * TILE_K +   16] = bReg1;

    int p = 0;
    for (int kk = 0; kk < K; kk += TILE_K) {
        __syncthreads();   // buf[p] ready; buf[1-p] fully consumed (prev iter)

        // ---- issue next tile's global loads early ----
        const bool more = (kk + TILE_K) < K;
        if (more) {
            const int gk = kk + TILE_K + aCol;
            const __bf16* src = (gk < K0) ? (A0 + (size_t)gmA * K0 + gk)
                                          : (A1 + (size_t)gmA * K1 + (gk - K0));
            aReg  = *(const bf16x16*)src;
            bReg0 = *(const bf16x16*)(Wt + (size_t)gnB * K + kk + TILE_K);
            bReg1 = *(const bf16x16*)(Wt + (size_t)gnB * K + kk + TILE_K + 16);
        }

        // ---- fragments from LDS buf[p] (CDNA5 16-bit A layout) ----
        v16bf afrag[2], bfrag[4];
        #pragma unroll
        for (int im = 0; im < 2; ++im) {
            const int r = wm * 32 + im * 16 + l15;
            bf16x16 u;
            u.q[0] = *(v4i*)&ldsA[p][r * TILE_K + half * 8];
            u.q[1] = *(v4i*)&ldsA[p][r * TILE_K + half * 8 + 16];
            afrag[im] = u.v;
        }
        #pragma unroll
        for (int in = 0; in < 4; ++in) {
            const int r = wn * 64 + in * 16 + l15;
            bf16x16 u;
            u.q[0] = *(v4i*)&ldsB[p][r * TILE_K + half * 8];
            u.q[1] = *(v4i*)&ldsB[p][r * TILE_K + half * 8 + 16];
            bfrag[in] = u.v;
        }

        #pragma unroll
        for (int im = 0; im < 2; ++im)
            #pragma unroll
            for (int in = 0; in < 4; ++in)
                acc[im][in] = __builtin_amdgcn_wmma_f32_16x16x32_bf16(
                    false, afrag[im], false, bfrag[in],
                    (short)0, acc[im][in], false, false);

        // ---- stage next tile into the other buffer ----
        if (more) {
            *(bf16x16*)&ldsA[1 - p][aRow * TILE_K + aCol] = aReg;
            *(bf16x16*)&ldsB[1 - p][bRow * TILE_K +    0] = bReg0;
            *(bf16x16*)&ldsB[1 - p][bRow * TILE_K +   16] = bReg1;
        }
        p ^= 1;
    }

    // ---- epilogue: bias + store (lane<16 -> M=r, lane>=16 -> M=r+8) ----
    #pragma unroll
    for (int im = 0; im < 2; ++im) {
        #pragma unroll
        for (int in = 0; in < 4; ++in) {
            const int n  = nTile + wn * 64 + in * 16 + l15;
            const float bv = bias ? bias[n] : 0.0f;
            const int mBase = mTile + wm * 32 + im * 16 + half * 8;
            #pragma unroll
            for (int r = 0; r < 8; ++r) {
                float v = acc[im][in][r] + bv;
                store_val(&out[(size_t)(mBase + r) * N + n], v);
            }
        }
    }
}

// ---------------------------------------------------------------------------
// LSTM elementwise cell: gates from lin (B x 4*NN fp32), update c, emit h (bf16)
// gate order per reference: f=[0:NN], i=[NN:2NN], o=[2NN:3NN], g=tanh([3NN:4NN])
// ---------------------------------------------------------------------------
__global__ void lstm_cell_kernel(const float* __restrict__ lin,
                                 float* __restrict__ c,
                                 __bf16* __restrict__ h_bf,
                                 __bf16* __restrict__ rnn_bf, // may be null
                                 int nb, int nn)
{
    int idx = blockIdx.x * blockDim.x + threadIdx.x;
    if (idx >= nb * nn) return;
    const int b = idx / nn;
    const int j = idx % nn;
    const float* row = lin + (size_t)b * 4 * nn;

    float fg = 1.0f / (1.0f + __expf(-row[j]));
    float ig = 1.0f / (1.0f + __expf(-row[nn + j]));
    float og = 1.0f / (1.0f + __expf(-row[2 * nn + j]));
    float g  = tanhf(row[3 * nn + j]);

    float cn = fg * c[idx] + ig * g;
    float hn = og * tanhf(cn);

    c[idx]    = cn;
    h_bf[idx] = f2bf(hn);
    if (rnn_bf) rnn_bf[idx] = f2bf(hn);
}

// ---------------------------------------------------------------------------
// Host-side orchestration
// ---------------------------------------------------------------------------
extern "C" void kernel_launch(void* const* d_in, const int* in_sizes, int n_in,
                              void* d_out, int out_size, void* d_ws, size_t ws_size,
                              hipStream_t stream) {
    const float* inputs = (const float*)d_in[0];   // T*B*V
    const float* emb    = (const float*)d_in[1];   // V*E
    const float* W0     = (const float*)d_in[2];   // (NN+E) x 4NN
    const float* b0     = (const float*)d_in[3];   // 4NN
    const float* W1     = (const float*)d_in[4];   // (2NN) x 4NN
    const float* b1     = (const float*)d_in[5];   // 4NN
    const float* Wout   = (const float*)d_in[6];   // NN x V
    const float* bout   = (const float*)d_in[7];   // V
    float* logits = (float*)d_out;                 // T*B x V

    const int TB  = T_ * B_;
    const int K0l = NN_ + E_;    // 1536
    const int K1l = 2 * NN_;     // 2048
    const int FN  = 4 * NN_;     // 4096

    // --- workspace carve-out ---
    char* ws = (char*)d_ws;
    size_t off = 0;
    auto carve = [&](size_t bytes) -> void* {
        off = (off + 255) & ~(size_t)255;
        void* p = ws + off;
        off += bytes;
        return p;
    };
    __bf16* Xbf     = (__bf16*)carve((size_t)TB * V_ * 2);        // inputs bf16
    __bf16* embT    = (__bf16*)carve((size_t)E_ * V_ * 2);        // E x V
    __bf16* W0T     = (__bf16*)carve((size_t)FN * K0l * 2);       // 4096 x 1536
    __bf16* W1T     = (__bf16*)carve((size_t)FN * K1l * 2);       // 4096 x 2048
    __bf16* WoutT   = (__bf16*)carve((size_t)V_ * NN_ * 2);       // 256 x 1024
    __bf16* embs_bf = (__bf16*)carve((size_t)TB * E_ * 2);        // x_t for all t
    __bf16* h0bf    = (__bf16*)carve((size_t)B_ * NN_ * 2);
    __bf16* h1bf    = (__bf16*)carve((size_t)B_ * NN_ * 2);
    float*  c0      = (float*) carve((size_t)B_ * NN_ * 4);
    float*  c1      = (float*) carve((size_t)B_ * NN_ * 4);
    float*  lin0    = (float*) carve((size_t)B_ * FN * 4);
    float*  lin1    = (float*) carve((size_t)B_ * FN * 4);
    __bf16* rnn_bf  = (__bf16*)carve((size_t)TB * NN_ * 2);
    (void)ws_size; (void)in_sizes; (void)n_in; (void)out_size;

    const int THR = 256;
    auto blocks = [&](long n) { return (int)((n + THR - 1) / THR); };

    // --- 1. conversions ---
    cvt_f32_to_bf16<<<blocks((long)TB * V_), THR, 0, stream>>>(inputs, Xbf, TB * V_);
    cvt_transpose_f32_to_bf16<<<blocks((long)V_ * E_), THR, 0, stream>>>(emb,  embT,  V_,  E_);
    cvt_transpose_f32_to_bf16<<<blocks((long)K0l * FN), THR, 0, stream>>>(W0,   W0T,   K0l, FN);
    cvt_transpose_f32_to_bf16<<<blocks((long)K1l * FN), THR, 0, stream>>>(W1,   W1T,   K1l, FN);
    cvt_transpose_f32_to_bf16<<<blocks((long)NN_ * V_), THR, 0, stream>>>(Wout, WoutT, NN_, V_);

    // --- 2. zero-init recurrent state ---
    zero_bytes_kernel<<<blocks((long)B_ * NN_ * 2), THR, 0, stream>>>((unsigned char*)h0bf, B_ * NN_ * 2);
    zero_bytes_kernel<<<blocks((long)B_ * NN_ * 2), THR, 0, stream>>>((unsigned char*)h1bf, B_ * NN_ * 2);
    zero_bytes_kernel<<<blocks((long)B_ * NN_ * 4), THR, 0, stream>>>((unsigned char*)c0,   B_ * NN_ * 4);
    zero_bytes_kernel<<<blocks((long)B_ * NN_ * 4), THR, 0, stream>>>((unsigned char*)c1,   B_ * NN_ * 4);

    // --- 3. embedding GEMM: embs = X @ emb  (M=32768, N=512, K=256), bf16 out ---
    {
        dim3 grid(E_ / TILE_N, TB / TILE_M);
        gemm_bf16_wmma<__bf16><<<grid, 128, 0, stream>>>(
            Xbf, V_, Xbf, 0, embT, nullptr, embs_bf, TB, E_);
    }

    // --- 4. recurrence over T timesteps ---
    dim3 gridStep(FN / TILE_N, B_ / TILE_M);   // 32 x 4 tiles
    const int cellBlocks = blocks((long)B_ * NN_);
    for (int t = 0; t < T_; ++t) {
        // layer 0: lin0 = [x_t | h0] @ W0 + b0
        gemm_bf16_wmma<float><<<gridStep, 128, 0, stream>>>(
            embs_bf + (size_t)t * B_ * E_, E_, h0bf, NN_, W0T, b0, lin0, B_, FN);
        lstm_cell_kernel<<<cellBlocks, THR, 0, stream>>>(lin0, c0, h0bf, nullptr, B_, NN_);
        // layer 1: lin1 = [h0 | h1] @ W1 + b1
        gemm_bf16_wmma<float><<<gridStep, 128, 0, stream>>>(
            h0bf, NN_, h1bf, NN_, W1T, b1, lin1, B_, FN);
        lstm_cell_kernel<<<cellBlocks, THR, 0, stream>>>(
            lin1, c1, h1bf, rnn_bf + (size_t)t * B_ * NN_, B_, NN_);
    }

    // --- 5. output GEMM: logits = rnn_out @ Wout + bout (M=32768, N=256, K=1024) ---
    {
        dim3 grid(V_ / TILE_N, TB / TILE_M);
        gemm_bf16_wmma<float><<<grid, 128, 0, stream>>>(
            rnn_bf, NN_, rnn_bf, 0, WoutT, bout, logits, TB, V_);
    }
}
